// Proposed_ver2_70815420776607
// MI455X (gfx1250) — matrix operand order repaired and checked by reading, not compiled, not run on hardware
//
#include <hip/hip_runtime.h>

typedef __attribute__((ext_vector_type(16))) __bf16 v16bf;
typedef __attribute__((ext_vector_type(8)))  __bf16 v8bf;
typedef __attribute__((ext_vector_type(8)))  float  v8f;
typedef __attribute__((ext_vector_type(4)))  float  f4;
typedef __attribute__((ext_vector_type(4)))  int    v4i_t;

#define HWDIM  256
#define GROUPS 16
#define NROWS  (256 * 512)          // N*C = 131072 rows of length 256
#define WPB    4                    // waves per block in routing kernel
#define EPSF   1e-5f

// ---- workspace layout (bytes) ----
#define WS_W1BF 0                   // 65536 bf16  = 131072 B
#define WS_W2BF 131072              // 4096  bf16  = 8192 B
#define WS_ACC  139264              // gsum[16], gss[16], gcnt[16] f32 = 192 B
#define WS_MEAN (139264 + 192)      // 16 f32
#define WS_RSTD (139264 + 256)      // 16 f32
#define WS_GID  140288              // 131072 u8

#if __has_builtin(__builtin_amdgcn_global_load_async_to_lds_b128)
#define USE_ASYNC_LDS 1
// builtin expects (v4i AS1* src, v4i AS3* dst, imm offset, imm cpol)
#define GA1(p) ((__attribute__((address_space(1))) v4i_t*)(p))
#define LA3(p) ((__attribute__((address_space(3))) v4i_t*)(p))
#endif

__device__ __forceinline__ void wait_async0() {
#if __has_builtin(__builtin_amdgcn_s_wait_asynccnt)
    __builtin_amdgcn_s_wait_asynccnt(0);
#else
    asm volatile("s_wait_asynccnt 0x0" ::: "memory");
#endif
}

// ---------------------------------------------------------------- K1: prep
__global__ __launch_bounds__(256) void prep_kernel(
    const float* __restrict__ w1, const float* __restrict__ w2,
    __bf16* __restrict__ w1bf, __bf16* __restrict__ w2bf,
    float* __restrict__ acc48)
{
    const int i = blockIdx.x * blockDim.x + threadIdx.x;
    if (i < 65536) w1bf[i] = (__bf16)w1[i];
    if (i < 4096)  w2bf[i] = (__bf16)w2[i];
    if (i < 48)    acc48[i] = 0.0f;
}

// ---------------------------------------------------------------- K2: routing MLP + stats
// Per wave: one 16-row tile of x.  GEMM1 B-fragments (w1) are staged chunk-by-chunk
// into LDS by the whole block (async DMA when available) and shared by all 4 waves,
// cutting L2 traffic 4x and turning B reads into ds_load_b128.
__global__ __launch_bounds__(128) void route_kernel(
    const float*  __restrict__ x,
    const __bf16* __restrict__ w1bf, const float* __restrict__ b1,
    const __bf16* __restrict__ w2bf, const float* __restrict__ b2,
    float* __restrict__ gsum, float* __restrict__ gss, float* __restrict__ gcnt,
    unsigned char* __restrict__ gid8)
{
    __shared__ __align__(16) __bf16 w1Lds[32 * HWDIM];         // 16 KB: 2 j-tiles of w1
    __shared__ __align__(16) __bf16 hidLds[WPB][16 * HWDIM];   // 32 KB hidden bounce
    __shared__ float lgt[WPB][16 * 16];                        // 4 KB logits
    __shared__ float red[WPB][2][32];                          // 1 KB row-stat partials
    __shared__ float bins[3][GROUPS];                          // per-block stat bins

    const int tid  = threadIdx.x;
    const int wv   = tid >> 5;          // wave id (wave32)
    const int lane = tid & 31;
    const int half = lane >> 4;         // lane group 0/1
    const int mrow = lane & 15;         // row (A) or column (B, C)

    if (tid < 3 * GROUPS) ((float*)bins)[tid] = 0.0f;

    const int tile    = blockIdx.x * WPB + wv;
    const int rowBase = tile * 16;

    // ---- A fragments: 16x256 tile of x, f32 -> bf16, per-row partial stats.
    // 16-bit A layout: lane holds row (lane&15), K = kb*32 + half*16 + {0..15}
    // in elements 0..15 (contiguous, matching B).
    v16bf a[8];
    float psum = 0.0f, pss = 0.0f;
    const float* xrow = x + (size_t)(rowBase + mrow) * HWDIM;
    #pragma unroll
    for (int kb = 0; kb < 8; ++kb) {
        const float* p = xrow + kb * 32 + half * 16;
        f4 c0 = *(const f4*)(p);
        f4 c1 = *(const f4*)(p + 4);
        f4 c2 = *(const f4*)(p + 8);
        f4 c3 = *(const f4*)(p + 12);
        #pragma unroll
        for (int t = 0; t < 4; ++t) {
            float f0 = c0[t], f1 = c1[t], f2 = c2[t], f3 = c3[t];
            a[kb][t]      = (__bf16)f0;
            a[kb][4 + t]  = (__bf16)f1;
            a[kb][8 + t]  = (__bf16)f2;
            a[kb][12 + t] = (__bf16)f3;
            psum += f0 + f1 + f2 + f3;
            pss  += f0 * f0 + f1 * f1 + f2 * f2 + f3 * f3;
        }
    }

    // ---- GEMM1: hidden[16x256] = x_tile @ w1^T + b1.
    // 8 chunks; each chunk stages 32 rows of w1 (2 n-tiles) into LDS for the block.
    for (int jc = 0; jc < 8; ++jc) {
        __syncthreads();   // all waves done reading previous chunk
        {
            // stage 32*256 bf16 = 16 KB: 128 threads x 8 x 16B pieces
            const __bf16* src = w1bf + (size_t)jc * 32 * HWDIM;
            #pragma unroll
            for (int t = 0; t < 8; ++t) {
                const unsigned idx = (unsigned)(tid * 8 + t) * 8u;  // element index
#ifdef USE_ASYNC_LDS
                __builtin_amdgcn_global_load_async_to_lds_b128(
                    GA1(src + idx), LA3(&w1Lds[idx]), 0, 0);
#else
                *(v8bf*)&w1Lds[idx] = *(const v8bf*)(src + idx);
#endif
            }
#ifdef USE_ASYNC_LDS
            wait_async0();
#endif
        }
        __syncthreads();   // staged chunk visible to all waves

        #pragma unroll
        for (int jj = 0; jj < 2; ++jj) {
            const int j = jc * 2 + jj;
            v8f acc = {};
            // B layout (32x16 bf16): lane holds column n = lane&15,
            // K = half*16 + {0..15}: 16 contiguous bf16 of w1 row n (ds_load_b128 x2).
            #pragma unroll
            for (int kb = 0; kb < 8; ++kb) {
                const v16bf bfrag = *(const v16bf*)
                    (&w1Lds[(jj * 16 + mrow) * HWDIM + kb * 32 + half * 16]);
                acc = __builtin_amdgcn_wmma_f32_16x16x32_bf16(
                    false, a[kb], false, bfrag, (short)0, acc, false, false);
            }
            const float bv = b1[j * 16 + mrow];
            // C layout: element v is row (v + 8*half), column mrow. Store bf16 row-major.
            #pragma unroll
            for (int v = 0; v < 8; ++v) {
                float h = acc[v] + bv;
                hidLds[wv][(v + 8 * half) * HWDIM + j * 16 + mrow] = (__bf16)h;
            }
        }
    }
    asm volatile("s_wait_dscnt 0x0" ::: "memory");   // in-wave LDS store -> cross-lane load

    // ---- GEMM2: logits[16x16] = hidden @ w2^T (reload hidden in A layout from LDS)
    v8f acc2 = {};
    #pragma unroll
    for (int kb = 0; kb < 8; ++kb) {
        const __bf16* hp = &hidLds[wv][mrow * HWDIM + kb * 32 + half * 16];
        v8bf lo = *(const v8bf*)(hp);
        v8bf hi = *(const v8bf*)(hp + 8);
        v16bf a2;
        #pragma unroll
        for (int t = 0; t < 8; ++t) { a2[t] = lo[t]; a2[8 + t] = hi[t]; }
        const v16bf b2f =
            *(const v16bf*)(w2bf + (size_t)mrow * HWDIM + kb * 32 + half * 16);
        acc2 = __builtin_amdgcn_wmma_f32_16x16x32_bf16(
            false, a2, false, b2f, (short)0, acc2, false, false);
    }
    const float b2v = b2[mrow];
    #pragma unroll
    for (int v = 0; v < 8; ++v)
        lgt[wv][(v + 8 * half) * 16 + mrow] = acc2[v] + b2v;

    red[wv][0][lane] = psum;
    red[wv][1][lane] = pss;
    asm volatile("s_wait_dscnt 0x0" ::: "memory");

    // ---- per-row argmax + stat accumulation (lanes 0..15 own one row each)
    if (lane < 16) {
        const int r = lane;
        float best = lgt[wv][r * 16];
        int   bi   = 0;
        #pragma unroll
        for (int g = 1; g < 16; ++g) {
            float v = lgt[wv][r * 16 + g];
            if (v > best) { best = v; bi = g; }   // first-max tie-break like argmax
        }
        const float rs = red[wv][0][r] + red[wv][0][r + 16];
        const float rq = red[wv][1][r] + red[wv][1][r + 16];
        gid8[rowBase + r] = (unsigned char)bi;
        atomicAdd(&bins[0][bi], rs);
        atomicAdd(&bins[1][bi], rq);
        atomicAdd(&bins[2][bi], 1.0f);
    }
    __syncthreads();
    if (tid < GROUPS) {
        atomicAdd(&gsum[tid], bins[0][tid]);
        atomicAdd(&gss[tid],  bins[1][tid]);
        atomicAdd(&gcnt[tid], bins[2][tid]);
    }
}

// ---------------------------------------------------------------- K3: finalize mean/rstd
__global__ void finalize_kernel(const float* __restrict__ gsum,
                                const float* __restrict__ gss,
                                const float* __restrict__ gcnt,
                                float* __restrict__ meanv,
                                float* __restrict__ rstdv)
{
    const int g = threadIdx.x;
    if (g < GROUPS) {
        const float cnt = gcnt[g] * (float)HWDIM;       // elements routed to group
        const float m   = gsum[g] / fmaxf(cnt, 1.0f);
        const float var = (gss[g] - cnt * m * m) / fmaxf(cnt - 1.0f, 1.0f); // ddof=1
        meanv[g] = m;
        rstdv[g] = rsqrtf(var + EPSF);
    }
}

// ---------------------------------------------------------------- K4: streaming normalize
__global__ __launch_bounds__(256) void norm_kernel(
    const float* __restrict__ x, const unsigned char* __restrict__ gid8,
    const float* __restrict__ meanv, const float* __restrict__ rstdv,
    const float* __restrict__ weight, const float* __restrict__ bias,
    float* __restrict__ out)
{
    const long long i4 = (long long)blockIdx.x * blockDim.x + threadIdx.x;
    const long long e  = i4 * 4;                 // 4 f32 per thread, exact cover
    const int row = (int)(e >> 8);               // 256 elements per row
    const int c   = row & 511;                   // channel index (C = 512)
    const int g   = gid8[row];
    const float m  = meanv[g];
    const float rs = rstdv[g];
    const float wv = weight[c];
    const float bv = bias[c];
    f4 xv = *(const f4*)(x + e);
    f4 o;
    #pragma unroll
    for (int t = 0; t < 4; ++t) o[t] = (xv[t] - m) * rs * wv + bv;
    *(f4*)(out + e) = o;
}

// ---------------------------------------------------------------- launcher
extern "C" void kernel_launch(void* const* d_in, const int* in_sizes, int n_in,
                              void* d_out, int out_size, void* d_ws, size_t ws_size,
                              hipStream_t stream)
{
    const float* x      = (const float*)d_in[0];
    const float* w1     = (const float*)d_in[1];
    const float* b1     = (const float*)d_in[2];
    const float* w2     = (const float*)d_in[3];
    const float* b2     = (const float*)d_in[4];
    const float* weight = (const float*)d_in[5];
    const float* bias   = (const float*)d_in[6];
    float* out = (float*)d_out;

    char* ws = (char*)d_ws;
    __bf16* w1bf = (__bf16*)(ws + WS_W1BF);
    __bf16* w2bf = (__bf16*)(ws + WS_W2BF);
    float*  acc  = (float*)(ws + WS_ACC);       // gsum | gss | gcnt (16 each)
    float*  gsum = acc;
    float*  gss  = acc + 16;
    float*  gcnt = acc + 32;
    float*  mean = (float*)(ws + WS_MEAN);
    float*  rstd = (float*)(ws + WS_RSTD);
    unsigned char* gid8 = (unsigned char*)(ws + WS_GID);

    // K1: weight conversion + accumulator reset (ws is poisoned by harness)
    prep_kernel<<<256, 256, 0, stream>>>(w1, w2, w1bf, w2bf, acc);

    // K2: 8192 row-tiles, 4 waves (tiles) per 128-thread block
    route_kernel<<<NROWS / 16 / WPB, WPB * 32, 0, stream>>>(
        x, w1bf, b1, w2bf, b2, gsum, gss, gcnt, gid8);

    // K3: finalize group statistics
    finalize_kernel<<<1, GROUPS, 0, stream>>>(gsum, gss, gcnt, mean, rstd);

    // K4: normalize + affine, float4 streaming (exactly covers 2^25 elements)
    norm_kernel<<<(NROWS * HWDIM) / 4 / 256, 256, 0, stream>>>(
        x, gid8, mean, rstd, weight, bias, out);
}
